// Net_SDE_11897059409951
// MI455X (gfx1250) — matrix operand order repaired and checked by reading, not compiled, not run on hardware
//
#include <hip/hip_runtime.h>
#include <math.h>

typedef __attribute__((ext_vector_type(16))) _Float16 v16h;
typedef __attribute__((ext_vector_type(8)))  _Float16 v8h;
typedef __attribute__((ext_vector_type(8)))  float    v8f;

#define NBLK   64
#define NTHR   128
#define NWAVE  4
#define NPATH  8192
#define NSTEP  48
#define NSLOT  128

// --- fast gfx1250 transcendentals (v_tanh_f32 is new on CDNA5) ---
#if defined(__has_builtin)
#if __has_builtin(__builtin_amdgcn_tanhf)
#define FAST_TANH(x) __builtin_amdgcn_tanhf(x)
#endif
#endif
#ifndef FAST_TANH
#define FAST_TANH(x) tanhf(x)
#endif

// ---------------- persistent device state (zero-init .bss) ----------------
__device__ __attribute__((aligned(16))) _Float16 gW[3][4][64 * 64];
__device__ float    gBias[3][4][64];
__device__ float    gWoutF[3][64];
__device__ __attribute__((aligned(16))) _Float16 gWcv[3][32 * 32];
__device__ float    gBcv[3][32];
__device__ __attribute__((aligned(16))) _Float16 gWcvOut[32 * 480];
__device__ float    g_cv[NPATH * 480];
__device__ float    g_part[NBLK * NSLOT];
__device__ unsigned g_cnt;
__device__ unsigned g_gen;

// ---------------- weight prep: f32 -> padded f16 ----------------
__global__ void prep_weights(const float* __restrict__ sp,
                             const float* __restrict__ ap,
                             const float* __restrict__ bp,
                             const float* __restrict__ cvp) {
  const int t      = blockIdx.x * blockDim.x + threadIdx.x;
  const int stride = gridDim.x * blockDim.x;
  const float* ps[3] = { sp, ap, bp };
  const int    dims[3] = { 3, 2, 2 };
  for (int net = 0; net < 3; ++net) {
    const float* p   = ps[net];
    const int    dim = dims[net];
    const float* W0  = p;
    const float* B0  = p + dim * 50;
    for (int i = t; i < 64 * 64; i += stride) {
      int k = i >> 6, n = i & 63;
      gW[net][0][i] = (_Float16)((k < dim && n < 50) ? W0[k * 50 + n] : 0.f);
    }
    for (int i = t; i < 64; i += stride) gBias[net][0][i] = (i < 50) ? B0[i] : 0.f;
    const float* q = p + dim * 50 + 50;
    for (int l = 1; l < 4; ++l) {
      const float* W = q;
      const float* B = q + 2500;
      for (int i = t; i < 64 * 64; i += stride) {
        int k = i >> 6, n = i & 63;
        gW[net][l][i] = (_Float16)((k < 50 && n < 50) ? W[k * 50 + n] : 0.f);
      }
      for (int i = t; i < 64; i += stride) gBias[net][l][i] = (i < 50) ? B[i] : 0.f;
      q += 2550;
    }
    for (int i = t; i < 64; i += stride) gWoutF[net][i] = (i < 50) ? q[i] : 0.f;
  }
  { // cv net: dim=2, width=30, 3 layers, out 30x480 (no output bias)
    const float* p = cvp;
    for (int i = t; i < 32 * 32; i += stride) {
      int k = i >> 5, n = i & 31;
      gWcv[0][i] = (_Float16)((k < 2 && n < 30) ? p[k * 30 + n] : 0.f);
    }
    const float* B0 = p + 60;
    for (int i = t; i < 32; i += stride) gBcv[0][i] = (i < 30) ? B0[i] : 0.f;
    const float* q = p + 90;
    for (int l = 1; l < 3; ++l) {
      for (int i = t; i < 32 * 32; i += stride) {
        int k = i >> 5, n = i & 31;
        gWcv[l][i] = (_Float16)((k < 30 && n < 30) ? q[k * 30 + n] : 0.f);
      }
      for (int i = t; i < 32; i += stride) gBcv[l][i] = (i < 30) ? q[900 + i] : 0.f;
      q += 930;
    }
    for (int i = t; i < 32 * 480; i += stride) {
      int k = i / 480, n = i % 480;
      gWcvOut[i] = (_Float16)((k < 30) ? q[k * 480 + n] : 0.f);
    }
  }
}

// ---------------- device helpers ----------------
__device__ __forceinline__ void grid_barrier() {
  __syncthreads();
  if (threadIdx.x == 0) {
    __threadfence();
    volatile unsigned* genp = &g_gen;
    unsigned gen = *genp;
    if (atomicAdd(&g_cnt, 1u) == (unsigned)(gridDim.x - 1)) {
      g_cnt = 0u;
      __threadfence();
      atomicAdd(&g_gen, 1u);
    } else {
      while (*genp == gen) { __builtin_amdgcn_s_sleep(2); }
    }
    __threadfence();
  }
  __syncthreads();
}

// deterministic block reduction -> g_part[block][slot]
__device__ __forceinline__ void breduce_store(float v, int slot, float* s_red) {
#pragma unroll
  for (int o = 16; o > 0; o >>= 1) v += __shfl_xor(v, o, 32);
  if ((threadIdx.x & 31) == 0) s_red[threadIdx.x >> 5] = v;
  __syncthreads();
  if (threadIdx.x == 0) {
    float s = 0.f;
#pragma unroll
    for (int w = 0; w < NWAVE; ++w) s += s_red[w];
    g_part[blockIdx.x * NSLOT + slot] = s;
  }
  __syncthreads();
}
// batched 6-value block reduction (payoff pass): one LDS round-trip
__device__ __forceinline__ void breduce6_store(const float* v, const int* slot, float* s_red6) {
  float r[6];
#pragma unroll
  for (int j = 0; j < 6; ++j) {
    float x = v[j];
#pragma unroll
    for (int o = 16; o > 0; o >>= 1) x += __shfl_xor(x, o, 32);
    r[j] = x;
  }
  if ((threadIdx.x & 31) == 0) {
    const int w = threadIdx.x >> 5;
#pragma unroll
    for (int j = 0; j < 6; ++j) s_red6[w * 6 + j] = r[j];
  }
  __syncthreads();
  if (threadIdx.x == 0) {
#pragma unroll
    for (int j = 0; j < 6; ++j) {
      float s = 0.f;
#pragma unroll
      for (int w = 0; w < NWAVE; ++w) s += s_red6[w * 6 + j];
      g_part[blockIdx.x * NSLOT + slot[j]] = s;
    }
  }
  __syncthreads();
}
// fixed-order global sum over block partials (deterministic)
__device__ __forceinline__ float gsum(int slot) {
  float s = 0.f;
  for (int b = 0; b < NBLK; ++b) s += g_part[b * NSLOT + slot];
  return s;
}
__device__ __forceinline__ float finsd(float sum, float sum2) {
  float mu  = sum * (1.f / 8192.f);
  float var = (sum2 - sum * mu) * (1.f / 8191.f);
  float sd  = sqrtf(fmaxf(var, 0.f));
  return (sd == 0.f) ? 1.f : sd;
}
// softplus via native v_exp_f32 / v_log_f32
__device__ __forceinline__ float softplus_f(float x) {
  return (x > 20.f) ? x : __logf(1.f + __expf(x));
}
__device__ __forceinline__ float softp05_f(float x) {
  float y = 0.5f * x;
  return (y > 20.f) ? x : 2.f * __logf(1.f + __expf(y));
}

// --- WMMA fragment loads (ISA 7.12.2 layouts, wave32) ---
__device__ __forceinline__ v16h load_A_frag(const _Float16* A, int mBase, int k0, int lane) {
  const int row = mBase + (lane & 15);
  const int kh  = (lane >> 4) * 8;
  const _Float16* p = A + row * 64 + k0 + kh;
  v8h lo = *(const v8h*)(p);
  v8h hi = *(const v8h*)(p + 16);
  v16h a;
#pragma unroll
  for (int e = 0; e < 8; ++e) { a[e] = lo[e]; a[8 + e] = hi[e]; }
  return a;
}
__device__ __forceinline__ v16h load_B_frag(const _Float16* W, int Npad, int k0, int nt, int lane) {
  const int n  = nt * 16 + (lane & 15);
  const int kh = (lane >> 4) * 8;
  const _Float16* p = W + (k0 + kh) * Npad + n;
  v16h b;
#pragma unroll
  for (int e = 0; e < 8; ++e) { b[e] = p[e * Npad]; b[8 + e] = p[(16 + e) * Npad]; }
  return b;
}

// one dense layer for a wave's 32 paths: tanh(A @ W + b) -> Aout (LDS, stride 64, f16)
template <int KPAD, int NPAD>
__device__ __forceinline__ void gemm_layer(const _Float16* A, const _Float16* W,
                                           const float* bias, _Float16* Aout, int lane) {
#pragma unroll
  for (int mt = 0; mt < 2; ++mt) {
#pragma unroll
    for (int nt = 0; nt < NPAD / 16; ++nt) {
      v8f c = {};
#pragma unroll
      for (int ks = 0; ks < KPAD / 32; ++ks) {
        v16h a = load_A_frag(A, mt * 16, ks * 32, lane);
        v16h b = load_B_frag(W, NPAD, ks * 32, nt, lane);
        c = __builtin_amdgcn_wmma_f32_16x16x32_f16(false, a, false, b, (short)0, c, false, false);
      }
      const int n  = nt * 16 + (lane & 15);
      const float bv = bias[n];
#pragma unroll
      for (int r = 0; r < 8; ++r) {
        const int m = mt * 16 + r + ((lane >> 4) << 3);
        Aout[m * 64 + n] = (_Float16)FAST_TANH(c[r] + bv);
      }
    }
  }
}

// full 4-layer width-50 MLP (padded 64); returns pre-activation scalar per lane's own path
__device__ __forceinline__ float run_net(int net, float x0, float x1, float x2, int dim,
                                         float skip, _Float16* A, _Float16* B2, int lane) {
#pragma unroll
  for (int j = 0; j < 32; ++j) {
    float v = (j == 0) ? x0 : (j == 1) ? x1 : ((j == 2 && dim == 3) ? x2 : 0.f);
    A[lane * 64 + j] = (_Float16)v;
  }
  __syncthreads();
  gemm_layer<32, 64>(A, gW[net][0], gBias[net][0], B2, lane);
  __syncthreads();
  gemm_layer<64, 64>(B2, gW[net][1], gBias[net][1], A, lane);
  __syncthreads();
  gemm_layer<64, 64>(A, gW[net][2], gBias[net][2], B2, lane);
  __syncthreads();
  gemm_layer<64, 64>(B2, gW[net][3], gBias[net][3], A, lane);
  __syncthreads();
  // output dot over padded width 64 (padded cols are exactly 0): vector LDS loads
  float acc = skip;
  const _Float16* hr = A + lane * 64;
  const float*    Wo = gWoutF[net];
#pragma unroll
  for (int c8 = 0; c8 < 8; ++c8) {
    v8h hv = *(const v8h*)(hr + c8 * 8);
#pragma unroll
    for (int e = 0; e < 8; ++e) acc += (float)hv[e] * Wo[c8 * 8 + e];
  }
  __syncthreads();
  return acc;
}

// ---------------- persistent SDE kernel (one launch, grid-barrier synced) ----------------
__launch_bounds__(NTHR, 1)
__global__ void sde_persistent(const float* __restrict__ rho_p,
                               const float* __restrict__ dW,
                               const float* __restrict__ dB,
                               float* __restrict__ out) {
  __shared__ __attribute__((aligned(16))) _Float16 s_act[NWAVE][2][32 * 64];
  __shared__ float s_red[NWAVE * 6];
  __shared__ float s_dvix[NTHR];
  __shared__ float s_skip[NTHR];

  const int tid  = threadIdx.x;
  const int lane = tid & 31;
  const int wave = tid >> 5;
  const int path = blockIdx.x * NTHR + tid;
  _Float16* A  = &s_act[wave][0][0];
  _Float16* B2 = &s_act[wave][1][0];

  // reset per-call state (harness replays this kernel; must be self-initializing)
  for (int j = 0; j < 480; ++j) g_cv[path * 480 + j] = 0.f;

  const float rho  = rho_p[0];
  const float corr = sqrtf(fmaxf(0.f, 1.f - rho * rho));
  const float Hs   = 1.f / 48.f;
  const float sqh  = sqrtf(Hs);
  const float DLT  = 4.f / 48.f; // VIX_DELTA

  float S = 1.f, V = 0.04f;

  for (int idx = 1; idx <= NSTEP; ++idx) {
    const float t = (float)(idx - 1) * Hs;

    // --- batch statistics of current S, V (normalize_input, ddof=1) ---
    breduce_store(S, 0, s_red);
    breduce_store(S * S, 1, s_red);
    breduce_store(V, 2, s_red);
    breduce_store(V * V, 3, s_red);
    grid_barrier();
    const float sumS = gsum(0), sumS2 = gsum(1), sumV = gsum(2), sumV2 = gsum(3);
    grid_barrier();
    const float muS = sumS * (1.f / 8192.f), sdS = finsd(sumS, sumS2);
    const float muV = sumV * (1.f / 8192.f), sdV = finsd(sumV, sumV2);
    const float Sn = (S - muS) / sdS;
    const float Vn = (V - muV) / sdV;

    // --- three MLPs (wave-collective WMMA GEMMs) ---
    const float sig  = softp05_f(run_net(0, t, Sn, Vn, 3, t + Sn + Vn, A, B2, lane));
    const float aout =            run_net(1, t, Vn, 0.f, 2, t + Vn,     A, B2, lane);
    const float bout = softplus_f(run_net(2, t, Vn, 0.f, 2, t + Vn,     A, B2, lane));

    // --- Euler step with taming + reflection ---
    const float dw = dW[path * NSTEP + (idx - 1)];
    const float db = dB[path * NSTEP + (idx - 1)];
    const float Z1 = S * sig;
    const float Snew = S * (1.f + 0.02f * Hs) + Z1 / (1.f + 2.5f * sqh * fabsf(Z1)) * dw;
    const float Vold = V;
    const float Z2 = aout * Hs + bout * (corr * db + rho * dw);
    const float Vnew = V + 0.05f * Z2 / fmaxf(1.f, Hs * fabsf(Z2));
    S = fabsf(Snew);
    V = fabsf(Vnew);

    // --- maturity processing ---
    if ((idx & 3) == 0) {
      const int it = (idx >> 2) - 1;
      const float disc = expf(-0.02f * t);
      const float vix  = sqrtf(fabsf(V    + 0.5f * aout * DLT)) * 100.f;
      const float vixo = sqrtf(fabsf(Vold + 0.5f * aout * DLT)) * 100.f;
      const float dv = vix - vixo;
      s_dvix[tid] = dv;

      // stats of new V (for cv normalization) + vix/V means
      breduce_store(V, 0, s_red);
      breduce_store(V * V, 1, s_red);
      breduce_store(vix, 2, s_red);
      grid_barrier();
      const float sV = gsum(0), sV2 = gsum(1), sVix = gsum(2);
      grid_barrier();
      if (blockIdx.x == 0 && tid == 0) {
        out[960 + it] = sV * (disc / 8192.f);   // fwd_var
        out[972 + it] = sVix * (disc / 8192.f); // vix_fwd
      }
      const float muV2 = sV * (1.f / 8192.f), sdV2 = finsd(sV, sV2);
      const float Vn2 = (V - muV2) / sdV2;
      s_skip[tid] = t + Vn2;
      __syncthreads();

      // cv MLP: 2->30->30->30 (padded 32), then 30->480 WMMA with cv accumulation
#pragma unroll
      for (int j = 0; j < 32; ++j)
        A[lane * 64 + j] = (_Float16)((j == 0) ? t : (j == 1) ? Vn2 : 0.f);
      __syncthreads();
      gemm_layer<32, 32>(A,  gWcv[0], gBcv[0], B2, lane);
      __syncthreads();
      gemm_layer<32, 32>(B2, gWcv[1], gBcv[1], A,  lane);
      __syncthreads();
      gemm_layer<32, 32>(A,  gWcv[2], gBcv[2], B2, lane);
      __syncthreads();
      const int pathBase = blockIdx.x * NTHR + wave * 32;
#pragma unroll
      for (int mt = 0; mt < 2; ++mt) {
        const v16h a0 = load_A_frag(B2, mt * 16, 0, lane);
        for (int nt = 0; nt < 30; ++nt) {
          const v16h bf = load_B_frag(gWcvOut, 480, 0, nt, lane);
          v8f c = {};
          c = __builtin_amdgcn_wmma_f32_16x16x32_f16(false, a0, false, bf, (short)0, c, false, false);
          const int n = nt * 16 + (lane & 15);
#pragma unroll
          for (int r = 0; r < 8; ++r) {
            const int ml = mt * 16 + r + ((lane >> 4) << 3);
            const int p  = pathBase + ml;
            g_cv[p * 480 + n] += (c[r] + s_skip[wave * 32 + ml]) * s_dvix[wave * 32 + ml];
          }
        }
      }
      __threadfence();
      __syncthreads();

      // payoff reductions: slots 0..119, batched 6 per strike
      for (int k = 0; k < 20; ++k) {
        const float strike = 0.8f + 0.4f * (float)k / 19.f;
        const float vk     = 10.f + 20.f * (float)k / 19.f;
        const float vc = (fmaxf(vix - vk, 0.f) - g_cv[path * 480 + it * 20 + k]) * disc;
        const float vp = (fmaxf(vk - vix, 0.f) - g_cv[path * 480 + (it + 12) * 20 + k]) * disc;
        const float vals[6] = { fmaxf(S - strike, 0.f) * disc, fmaxf(strike - S, 0.f) * disc,
                                vc, vp, vc * vc, vp * vp };
        const int slots[6] = { k, 20 + k, 40 + k, 60 + k, 80 + k, 100 + k };
        breduce6_store(vals, slots, s_red);
      }
      grid_barrier();
      if (blockIdx.x == 0 && tid < 20) {
        const int k = tid;
        const float eqc = gsum(k), eqp = gsum(20 + k);
        const float vc = gsum(40 + k), vp = gsum(60 + k);
        const float vc2 = gsum(80 + k), vp2 = gsum(100 + k);
        out[it * 20 + k]              = eqc * (1.f / 8192.f);
        out[(it + 12) * 20 + k]       = eqp * (1.f / 8192.f);
        out[480 + it * 20 + k]        = vc * (1.f / 8192.f);
        out[480 + (it + 12) * 20 + k] = vp * (1.f / 8192.f);
        const float mc = vc * (1.f / 8192.f), mp = vp * (1.f / 8192.f);
        out[984 + it * 20 + k]        = (vc2 - 8192.f * mc * mc) * (1.f / 8191.f);
        out[984 + (it + 12) * 20 + k] = (vp2 - 8192.f * mp * mp) * (1.f / 8191.f);
      }
      grid_barrier();
    }
  }
}

extern "C" void kernel_launch(void* const* d_in, const int* in_sizes, int n_in,
                              void* d_out, int out_size, void* d_ws, size_t ws_size,
                              hipStream_t stream) {
  (void)in_sizes; (void)n_in; (void)d_ws; (void)ws_size; (void)out_size;
  const float* sigmaP = (const float*)d_in[0];
  const float* aP     = (const float*)d_in[1];
  const float* bP     = (const float*)d_in[2];
  const float* cvP    = (const float*)d_in[3];
  const float* rho    = (const float*)d_in[4];
  const float* dW     = (const float*)d_in[5];
  const float* dB     = (const float*)d_in[6];
  float* out = (float*)d_out;

  prep_weights<<<dim3(16), dim3(256), 0, stream>>>(sigmaP, aP, bP, cvP);
  sde_persistent<<<dim3(NBLK), dim3(NTHR), 0, stream>>>(rho, dW, dB, out);
}
